// CWVAE_81123342287083
// MI455X (gfx1250) — matrix operand order repaired and checked by reading, not compiled, not run on hardware
//
#include <hip/hip_runtime.h>
#include <hip/hip_bf16.h>

// ---------------------------------------------------------------------------
// CWVAE on MI455X (gfx1250): persistent per-level GRU scan with bf16 WMMA.
// Batch split across 2 workgroups (16 rows each); all state in LDS; weights
// pre-packed to WMMA B-fragment layout in workspace (bf16, resident in L2).
// ---------------------------------------------------------------------------

typedef __bf16 bf16_t;
typedef __attribute__((ext_vector_type(16))) __bf16 v16bf;
typedef __attribute__((ext_vector_type(8)))  float  v8f;
typedef __attribute__((ext_vector_type(4)))  unsigned int u32x4;

#define STOCH   64
#define DETER   512
#define EMBED   512
#define OBSE    1024
#define XIN_K   (STOCH + DETER)   // 576
#define HOBS_K  (DETER + OBSE)    // 1536
#define ROWS    16                // batch rows per workgroup (1 WMMA M-tile)
#define NTHREADS 512
#define NWAVES  16

union FragU { u32x4 u[2]; v16bf v; };

// A operand 16x32 bf16 from row-major LDS activations.
// ISA layout: lanes 0-15 (M=lane): V0-3 K=kb+0..7, V4-7 K=kb+16..23;
//             lanes 16-31 (M=lane-16): K=kb+8..15 / kb+24..31.
__device__ __forceinline__ v16bf load_a_frag(const bf16_t* lds, int stride, int kb, int lane) {
  const int m    = lane & 15;
  const int koff = (lane >> 4) << 3;
  const bf16_t* p = lds + m * stride + kb + koff;
  FragU f;
  f.u[0] = *(const u32x4*)(p);        // 16B: K koff..koff+7
  f.u[1] = *(const u32x4*)(p + 16);   // 16B: K 16+koff..23+koff
  return f.v;
}

// B operand 32x16 bf16 from pre-packed fragment layout: per (nt,kt) block,
// lane holds 16 contiguous bf16 (32 bytes). Prefetch next K-block.
__device__ __forceinline__ v16bf load_b_frag(const bf16_t* w, int nbk, int nt, int kt, int lane) {
  const bf16_t* p = w + (size_t)(((nt * nbk + kt) << 5) + lane) * 16;
  __builtin_prefetch(p + 512, 0, 1);  // next kt block -> global_prefetch_b8
  FragU f;
  f.u[0] = *(const u32x4*)(p);
  f.u[1] = *(const u32x4*)(p + 8);
  return f.v;
}

__device__ __forceinline__ v8f wmma_bf16(v16bf a, v16bf b, v8f c) {
  return __builtin_amdgcn_wmma_f32_16x16x32_bf16(false, a, false, b, (short)0, c, false, false);
}

__device__ __forceinline__ float sigmoidf_(float x) { return 1.0f / (1.0f + __expf(-x)); }

// ---------------------------------------------------------------------------
// Pack fp32 [K x N] (row stride srcLd) into bf16 WMMA B-fragment blocks.
// Block (nt,kt): lane L holds K = kt*32 + (L>>4)*8 + (idx&7) + (idx>>3)*16, N = nt*16 + (L&15).
// ---------------------------------------------------------------------------
__global__ void pack_weights_b(const float* __restrict__ src, bf16_t* __restrict__ dst,
                               int nbk, int srcLd, int total) {
  for (int i = blockIdx.x * blockDim.x + threadIdx.x; i < total;
       i += gridDim.x * blockDim.x) {
    const int idx  = i & 15;
    const int lane = (i >> 4) & 31;
    const int blk  = i >> 9;
    const int kt   = blk % nbk;
    const int nt   = blk / nbk;
    const int k = (kt << 5) + ((lane >> 4) << 3) + (idx & 7) + ((idx >> 3) << 4);
    const int n = (nt << 4) + (lane & 15);
    dst[i] = (bf16_t)src[(size_t)k * srcLd + n];
  }
}

// ---------------------------------------------------------------------------
// One level's full time scan. grid = 2 workgroups (16 batch rows each).
// ---------------------------------------------------------------------------
__global__ __launch_bounds__(NTHREADS, 1)
void cwvae_level(const bf16_t* __restrict__ Win,  const bf16_t* __restrict__ Wih,
                 const bf16_t* __restrict__ Whh,  const bf16_t* __restrict__ Wobs,
                 const bf16_t* __restrict__ Wpost,
                 const float* __restrict__ b_in,  const float* __restrict__ b_ih,
                 const float* __restrict__ b_hh,  const float* __restrict__ b_obs,
                 const float* __restrict__ b_post,
                 const float* __restrict__ obs,   const float* __restrict__ ctx,
                 float* __restrict__ det, int T, int ctxT)
{
  extern __shared__ __align__(16) char smem_raw[];
  bf16_t* s_xin  = (bf16_t*)smem_raw;            // ROWS x 576  [sample | ctx]
  bf16_t* s_x    = s_xin  + ROWS * XIN_K;        // ROWS x 512
  bf16_t* s_h    = s_x    + ROWS * EMBED;        // ROWS x 512  (committed h)
  bf16_t* s_hobs = s_h    + ROWS * DETER;        // ROWS x 1536 [h_stage | obs]
  bf16_t* s_hx   = s_hobs + ROWS * HOBS_K;       // ROWS x 512

  const int tid  = threadIdx.x;
  const int lane = tid & 31;
  const int w    = tid >> 5;
  const int brow = blockIdx.x * ROWS;

  // init: h = 0, sample = 0, staged-h = 0
  for (int i = tid; i < ROWS * DETER; i += NTHREADS) {
    s_h[i] = (bf16_t)0.f;
    s_hobs[(i >> 9) * HOBS_K + (i & 511)] = (bf16_t)0.f;
  }
  for (int i = tid; i < ROWS * STOCH; i += NTHREADS)
    s_xin[(i >> 6) * XIN_K + (i & 63)] = (bf16_t)0.f;
  __syncthreads();

  for (int t = 0; t < T; ++t) {
    // ---- phase 0: stream ctx (upper-level det, time-repeated x4) and obs
    for (int i = tid; i < ROWS * DETER; i += NTHREADS) {
      const int r = i >> 9, c = i & 511;
      const float v = ctx ? ctx[((size_t)(brow + r) * ctxT + (t >> 2)) * DETER + c] : 0.0f;
      s_xin[r * XIN_K + STOCH + c] = (bf16_t)v;
    }
    for (int i = tid; i < ROWS * OBSE; i += NTHREADS) {
      const int r = i >> 10, c = i & 1023;
      const float v = obs[((size_t)(brow + r) * T + t) * OBSE + c];
      s_hobs[r * HOBS_K + DETER + c] = (bf16_t)v;
    }
    __syncthreads();

    // ---- phase 1: x = relu([sample,ctx] @ Win + b_in)   K=576 (18 kt), N=512
    for (int nt = w; nt < 32; nt += NWAVES) {
      v8f acc = {};
      for (int kt = 0; kt < 18; ++kt) {
        const v16bf a = load_a_frag(s_xin, XIN_K, kt << 5, lane);
        const v16bf b = load_b_frag(Win, 18, nt, kt, lane);
        acc = wmma_bf16(a, b, acc);
      }
      const int   n    = (nt << 4) + (lane & 15);
      const float bias = b_in[n];
      const int   mb   = (lane >> 4) << 3;
      #pragma unroll
      for (int v = 0; v < 8; ++v) {
        const float xv = acc[v] + bias;
        s_x[(mb + v) * EMBED + n] = (bf16_t)(xv > 0.f ? xv : 0.f);
      }
    }
    __syncthreads();

    // ---- phase 2: fused GRU. Each wave owns a 16-col h-tile; 6 accumulators.
    for (int nt = w; nt < 32; nt += NWAVES) {
      v8f air = {}, aiz = {}, ain = {}, ahr = {}, ahz = {}, ahn = {};
      for (int kt = 0; kt < 16; ++kt) {
        const v16bf ax = load_a_frag(s_x, EMBED, kt << 5, lane);
        const v16bf ah = load_a_frag(s_h, DETER, kt << 5, lane);
        air = wmma_bf16(ax, load_b_frag(Wih, 16, nt,      kt, lane), air);
        aiz = wmma_bf16(ax, load_b_frag(Wih, 16, nt + 32, kt, lane), aiz);
        ain = wmma_bf16(ax, load_b_frag(Wih, 16, nt + 64, kt, lane), ain);
        ahr = wmma_bf16(ah, load_b_frag(Whh, 16, nt,      kt, lane), ahr);
        ahz = wmma_bf16(ah, load_b_frag(Whh, 16, nt + 32, kt, lane), ahz);
        ahn = wmma_bf16(ah, load_b_frag(Whh, 16, nt + 64, kt, lane), ahn);
      }
      const int   n    = (nt << 4) + (lane & 15);
      const float bir = b_ih[n], biz = b_ih[512 + n], bin_ = b_ih[1024 + n];
      const float bhr = b_hh[n], bhz = b_hh[512 + n], bhn  = b_hh[1024 + n];
      const int   mb  = (lane >> 4) << 3;
      #pragma unroll
      for (int v = 0; v < 8; ++v) {
        const int   m    = mb + v;
        const float r    = sigmoidf_(air[v] + bir + ahr[v] + bhr);
        const float z    = sigmoidf_(aiz[v] + biz + ahz[v] + bhz);
        const float nn   = tanhf(ain[v] + bin_ + r * (ahn[v] + bhn));
        const float hp   = (float)s_h[m * DETER + n];
        const float hnew = (1.f - z) * nn + z * hp;
        s_hobs[m * HOBS_K + n] = (bf16_t)hnew;  // stage (s_h still being read)
        det[((size_t)(brow + m) * T + t) * DETER + n] = hnew;
      }
    }
    __syncthreads();

    // ---- phase 2b: commit staged h
    for (int i = tid; i < ROWS * DETER; i += NTHREADS)
      s_h[i] = s_hobs[(i >> 9) * HOBS_K + (i & 511)];
    __syncthreads();

    // ---- phase 3: hx = relu([h,obs] @ Wobs + b_obs)   K=1536 (48 kt), N=512
    for (int nt = w; nt < 32; nt += NWAVES) {
      v8f acc = {};
      for (int kt = 0; kt < 48; ++kt) {
        const v16bf a = load_a_frag(s_hobs, HOBS_K, kt << 5, lane);
        const v16bf b = load_b_frag(Wobs, 48, nt, kt, lane);
        acc = wmma_bf16(a, b, acc);
      }
      const int   n    = (nt << 4) + (lane & 15);
      const float bias = b_obs[n];
      const int   mb   = (lane >> 4) << 3;
      #pragma unroll
      for (int v = 0; v < 8; ++v) {
        const float xv = acc[v] + bias;
        s_hx[(mb + v) * EMBED + n] = (bf16_t)(xv > 0.f ? xv : 0.f);
      }
    }
    __syncthreads();

    // ---- phase 4: sample = posterior mean = hx @ Wpost[:, :64] + b_post[:64]
    if (w < 4) {
      const int nt = w;
      v8f acc = {};
      for (int kt = 0; kt < 16; ++kt) {
        const v16bf a = load_a_frag(s_hx, EMBED, kt << 5, lane);
        const v16bf b = load_b_frag(Wpost, 16, nt, kt, lane);
        acc = wmma_bf16(a, b, acc);
      }
      const int   n    = (nt << 4) + (lane & 15);
      const float bias = b_post[n];
      const int   mb   = (lane >> 4) << 3;
      #pragma unroll
      for (int v = 0; v < 8; ++v)
        s_xin[(mb + v) * XIN_K + n] = (bf16_t)(acc[v] + bias);
    }
    __syncthreads();
  }
}

// ---------------------------------------------------------------------------
extern "C" void kernel_launch(void* const* d_in, const int* in_sizes, int n_in,
                              void* d_out, int out_size, void* d_ws, size_t ws_size,
                              hipStream_t stream) {
  (void)in_sizes; (void)n_in; (void)out_size; (void)ws_size;
  const float* obs0 = (const float*)d_in[0];
  const float* obs1 = (const float*)d_in[1];
  const float* obs2 = (const float*)d_in[2];
  // params: 12 tensors per level, dict order:
  // 0 W_in, 1 b_in, 2 W_ih, 3 b_ih, 4 W_hh, 5 b_hh,
  // 6 W_prior, 7 b_prior, 8 W_obs, 9 b_obs, 10 W_post, 11 b_post
  auto Pw = [&](int lvl, int j) -> const float* {
    return (const float*)d_in[3 + lvl * 12 + j];
  };

  // workspace carve-out
  char* ws = (char*)d_ws;
  size_t off = 0;
  auto carve = [&](size_t bytes) -> void* {
    void* p = ws + off;
    off += (bytes + 255) & ~(size_t)255;
    return p;
  };

  const size_t nWin = 576 * 512, nWih = 512 * 1536, nWhh = 512 * 1536,
               nWobs = 1536 * 512, nWpost = 512 * 64;
  bf16_t *pWin[3], *pWih[3], *pWhh[3], *pWobs[3], *pWpost[3];
  for (int l = 0; l < 3; ++l) {
    pWin[l]   = (bf16_t*)carve(nWin   * 2);
    pWih[l]   = (bf16_t*)carve(nWih   * 2);
    pWhh[l]   = (bf16_t*)carve(nWhh   * 2);
    pWobs[l]  = (bf16_t*)carve(nWobs  * 2);
    pWpost[l] = (bf16_t*)carve(nWpost * 2);
  }
  float* det2 = (float*)carve((size_t)32 * 64  * 512 * 4);
  float* det1 = (float*)carve((size_t)32 * 256 * 512 * 4);

  auto pack = [&](const float* src, bf16_t* dst, int K, int N, int srcLd) {
    const int total = K * N;
    int grid = (total + 255) / 256;
    if (grid > 4096) grid = 4096;
    pack_weights_b<<<dim3(grid), dim3(256), 0, stream>>>(src, dst, K / 32, srcLd, total);
  };
  for (int l = 0; l < 3; ++l) {
    pack(Pw(l, 0),  pWin[l],   576,  512,  512);
    pack(Pw(l, 2),  pWih[l],   512,  1536, 1536);
    pack(Pw(l, 4),  pWhh[l],   512,  1536, 1536);
    pack(Pw(l, 8),  pWobs[l],  1536, 512,  512);
    pack(Pw(l, 10), pWpost[l], 512,  64,   128);   // only posterior-mean cols
  }

  const size_t smem = (size_t)(ROWS * XIN_K + ROWS * EMBED + ROWS * DETER +
                               ROWS * HOBS_K + ROWS * EMBED) * sizeof(bf16_t); // 114 KB

  // top level (2): zero context, T=64
  cwvae_level<<<dim3(2), dim3(NTHREADS), smem, stream>>>(
      pWin[2], pWih[2], pWhh[2], pWobs[2], pWpost[2],
      Pw(2, 1), Pw(2, 3), Pw(2, 5), Pw(2, 9), Pw(2, 11),
      obs2, (const float*)nullptr, det2, 64, 0);
  // level 1: ctx = det2 repeated x4, T=256
  cwvae_level<<<dim3(2), dim3(NTHREADS), smem, stream>>>(
      pWin[1], pWih[1], pWhh[1], pWobs[1], pWpost[1],
      Pw(1, 1), Pw(1, 3), Pw(1, 5), Pw(1, 9), Pw(1, 11),
      obs1, det2, det1, 256, 64);
  // level 0: ctx = det1 repeated x4, T=1024, det -> final output
  cwvae_level<<<dim3(2), dim3(NTHREADS), smem, stream>>>(
      pWin[0], pWih[0], pWhh[0], pWobs[0], pWpost[0],
      Pw(0, 1), Pw(0, 3), Pw(0, 5), Pw(0, 9), Pw(0, 11),
      obs0, det1, (float*)d_out, 1024, 256);
}
